// EquiGraph2GraphModel_47768626266284
// MI455X (gfx1250) — compile-verified, matrix operand order checked by reading, not run on hardware
//
#include <hip/hip_runtime.h>
#include <math.h>

#define E_N 262144
#define N_N 16384
#define G_N 64
#define DTP 108

typedef __attribute__((ext_vector_type(16))) __bf16 v16bf;
typedef __attribute__((ext_vector_type(8)))  __bf16 v8bf;
typedef __attribute__((ext_vector_type(8)))  float  v8f;

// ---------------------------------------------------------------------------
// Generic gather-GEMM:  C = postscale * act( alpha * (concat(segs) @ W) + b )
// A rows are gathered/concatenated from up to 4 float32 segments (optional
// row-index arrays), converted to bf16 in LDS, and multiplied against a
// pre-packed bf16 weight matrix WT[Npad][Kpad] (= W transposed, zero padded)
// with v_wmma_f32_16x16x32_bf16.  One wave per 16x64 output tile:
// 4 accumulators / 4 WMMAs per K-step so the gathered A fragment is reused 4x.
// ---------------------------------------------------------------------------
struct Seg { const float* ptr; const int* idx; int width; };
struct GemmP {
  Seg seg[4]; int nseg;
  long long M; int K, Kpad, N, Npad;       // Npad: multiple of 64
  const __bf16* WT; const float* bias;
  float alpha, postscale; int relu;
  float* C; int ldC; int colOff;
};

__global__ __launch_bounds__(256) void gemm_bf16_wmma(GemmP p) {
  // per wave: A tile 16x32 (512) + 4 B subtiles 16x32 (2048) = 2560 bf16
  __shared__ __attribute__((aligned(16))) __bf16 lds[8][2560];
  const int wave = threadIdx.x >> 5;
  const int t    = threadIdx.x & 31;
  const int ntiles = p.Npad >> 6;
  const long long T = (p.M >> 4) * (long long)ntiles;
  long long tile = (long long)blockIdx.x * 8 + wave;
  const bool active = tile < T;
  if (!active) tile = 0;
  const long long m0 = (tile / ntiles) << 4;
  const int n0 = (int)(tile % ntiles) << 6;
  __bf16* As = lds[wave];
  __bf16* Bs = As + 512;
  const int r  = t >> 1, ch = (t & 1) << 4;   // tile fill: row, k-half
  const int lr = t & 15, g  = t >> 4;         // fragment: lane-row, half

  // Hoist per-row segment resolution out of the K loop: this lane always
  // feeds A row (m0 + r); idx[] gathers happen exactly once.
  const long long m = m0 + r;
  const float* rb0; const float* rb1; const float* rb2; const float* rb3;
  int en0, en1, en2; int st1, st2, st3;
  {
    const Seg s0 = p.seg[0];
    long long ri0 = s0.idx ? (long long)s0.idx[m] : m;
    rb0 = s0.ptr + ri0 * s0.width; en0 = s0.width;
    rb1 = rb0; rb2 = rb0; rb3 = rb0;
    st1 = 0; st2 = 0; st3 = 0;
    en1 = 0x7fffffff; en2 = 0x7fffffff;
    int cum = s0.width;
    if (p.nseg > 1) {
      const Seg s1 = p.seg[1];
      long long ri = s1.idx ? (long long)s1.idx[m] : m;
      rb1 = s1.ptr + ri * s1.width; st1 = cum; cum += s1.width; en1 = cum;
      rb2 = rb1; rb3 = rb1; st2 = st1; st3 = st1;
    }
    if (p.nseg > 2) {
      const Seg s2 = p.seg[2];
      long long ri = s2.idx ? (long long)s2.idx[m] : m;
      rb2 = s2.ptr + ri * s2.width; st2 = cum; cum += s2.width; en2 = cum;
      rb3 = rb2; st3 = st2;
    }
    if (p.nseg > 3) {
      const Seg s3 = p.seg[3];
      long long ri = s3.idx ? (long long)s3.idx[m] : m;
      rb3 = s3.ptr + ri * s3.width; st3 = cum;
    }
  }
  // Fast path: single contiguous segment with no padding (all l2/fc2 GEMMs).
  const bool fastA = (p.nseg == 1) && (p.seg[0].idx == nullptr) && (p.Kpad == p.K);

  v8f acc0 = {0.f,0.f,0.f,0.f,0.f,0.f,0.f,0.f};
  v8f acc1 = acc0, acc2 = acc0, acc3 = acc0;

  union { __bf16 h[16]; uint4 q[2]; } tmp;

  for (int k0 = 0; k0 < p.Kpad; k0 += 32) {
    // ---- A tile: gather + convert (each lane: 1 row x 16 contiguous k) ----
    if (fastA) {
      const float4* s = reinterpret_cast<const float4*>(rb0 + k0 + ch);
      float4 f0 = s[0], f1 = s[1], f2 = s[2], f3 = s[3];
      float fb[16] = {f0.x,f0.y,f0.z,f0.w, f1.x,f1.y,f1.z,f1.w,
                      f2.x,f2.y,f2.z,f2.w, f3.x,f3.y,f3.z,f3.w};
      #pragma unroll
      for (int j = 0; j < 16; ++j) tmp.h[j] = (__bf16)fb[j];
    } else {
      #pragma unroll
      for (int j = 0; j < 16; ++j) {
        const int f = k0 + ch + j;
        const float* q = rb0; int o = f;
        if (f >= en0) { q = rb1; o = f - st1; }
        if (f >= en1) { q = rb2; o = f - st2; }
        if (f >= en2) { q = rb3; o = f - st3; }
        float val = (f < p.K) ? q[o] : 0.f;
        tmp.h[j] = (__bf16)val;
      }
    }
    *reinterpret_cast<uint4*>(As + r * 32 + ch)     = tmp.q[0];
    *reinterpret_cast<uint4*>(As + r * 32 + ch + 8) = tmp.q[1];
    // ---- B tiles: 4 subtiles of 16 n-rows x 32 k from packed WT ----
    #pragma unroll
    for (int sub = 0; sub < 4; ++sub) {
      const __bf16* src = p.WT + (size_t)(n0 + sub * 16 + r) * p.Kpad + k0 + ch;
      __builtin_prefetch(src + 32, 0, 0);      // global_prefetch_b8
      const uint4* s4 = reinterpret_cast<const uint4*>(src);
      *reinterpret_cast<uint4*>(Bs + sub * 512 + r * 32 + ch)     = s4[0];
      *reinterpret_cast<uint4*>(Bs + sub * 512 + r * 32 + ch + 8) = s4[1];
    }
    __syncthreads();
    // Fragments per CDNA5 16-bit layout: lanes 0-15 hold K {g*8..g*8+7} and
    // {16+g*8..}, contiguous in our row-major LDS tiles -> b128 loads.
    v8bf alo = *reinterpret_cast<const v8bf*>(As + lr * 32 + g * 8);
    v8bf ahi = *reinterpret_cast<const v8bf*>(As + lr * 32 + g * 8 + 16);
    v16bf av = __builtin_shufflevector(alo, ahi, 0,1,2,3,4,5,6,7,8,9,10,11,12,13,14,15);
    #pragma unroll
    for (int sub = 0; sub < 4; ++sub) {
      v8bf blo = *reinterpret_cast<const v8bf*>(Bs + sub * 512 + lr * 32 + g * 8);
      v8bf bhi = *reinterpret_cast<const v8bf*>(Bs + sub * 512 + lr * 32 + g * 8 + 16);
      v16bf bv = __builtin_shufflevector(blo, bhi, 0,1,2,3,4,5,6,7,8,9,10,11,12,13,14,15);
      if (sub == 0) acc0 = __builtin_amdgcn_wmma_f32_16x16x32_bf16(false, av, false, bv, (short)0, acc0, false, false);
      if (sub == 1) acc1 = __builtin_amdgcn_wmma_f32_16x16x32_bf16(false, av, false, bv, (short)0, acc1, false, false);
      if (sub == 2) acc2 = __builtin_amdgcn_wmma_f32_16x16x32_bf16(false, av, false, bv, (short)0, acc2, false, false);
      if (sub == 3) acc3 = __builtin_amdgcn_wmma_f32_16x16x32_bf16(false, av, false, bv, (short)0, acc3, false, false);
    }
    __syncthreads();
  }
  if (active) {
    #pragma unroll
    for (int sub = 0; sub < 4; ++sub) {
      const v8f acc = (sub == 0) ? acc0 : (sub == 1) ? acc1 : (sub == 2) ? acc2 : acc3;
      const int n = n0 + sub * 16 + lr;
      if (n < p.N) {
        const float b = p.bias ? p.bias[n] : 0.f;
        #pragma unroll
        for (int v = 0; v < 8; ++v) {          // C layout: M = v + 8*g, N = lr
          const long long mm = m0 + g * 8 + v;
          if (mm < p.M) {
            float val = acc[v] * p.alpha + b;
            if (p.relu) val = fmaxf(val, 0.f);
            p.C[mm * (long long)p.ldC + p.colOff + n] = val * p.postscale;
          }
        }
      }
    }
  }
}

// ---------------------------------------------------------------------------
// Weight packing: W (KxN row-major f32) -> WT (Npad x Kpad bf16), rows=N cols
// (zero padding comes from a prior memset of the whole weight region).
// ---------------------------------------------------------------------------
__global__ void pack_wt(const float* W, __bf16* WT, int K, int N, int Kpad) {
  int i = blockIdx.x * blockDim.x + threadIdx.x;
  if (i >= N * Kpad) return;
  int n = i / Kpad, k = i % Kpad;
  float v = (k < K) ? W[(size_t)k * N + n] : 0.f;
  WT[(size_t)n * Kpad + k] = (__bf16)v;
}

// ---------------------------------------------------------------------------
// Wigner-3j tables computed on device (replicates _su2_cg/_qmat/_wigner3j).
// ---------------------------------------------------------------------------
__device__ double dfact(int n) { double r = 1.0; for (int i = 2; i <= n; ++i) r *= i; return r; }
__device__ double su2cg(int j1,int m1,int j2,int m2,int j3,int m3) {
  if (m3 != m1 + m2) return 0.0;
  int vmin = -j1 + j2 + m3; if (-j1 + m1 > vmin) vmin = -j1 + m1; if (0 > vmin) vmin = 0;
  int vmax = j2 + j3 + m1; if (j3 - j1 + j2 < vmax) vmax = j3 - j1 + j2; if (j3 + m3 < vmax) vmax = j3 + m3;
  double C = sqrt((2.0*j3+1.0)*dfact(j3+j1-j2)*dfact(j3-j1+j2)*dfact(j1+j2-j3)*dfact(j3+m3)*dfact(j3-m3)
             /(dfact(j1+j2+j3+1)*dfact(j1-m1)*dfact(j1+m1)*dfact(j2-m2)*dfact(j2+m2)));
  double S = 0.0;
  for (int v = vmin; v <= vmax; ++v) {
    double sg = ((v + j2 + m2) & 1) ? -1.0 : 1.0;
    S += sg * dfact(j2+j3+m1-v)*dfact(j1-m1+v)
       / (dfact(v)*dfact(j3-j1+j2-v)*dfact(j3+m3-v)*dfact(v+j1-j2-m3));
  }
  return C * S;
}
__device__ void qmat(int l, double qr[5][5], double qi[5][5]) {
  for (int a = 0; a < 5; ++a) for (int b = 0; b < 5; ++b) { qr[a][b] = 0; qi[a][b] = 0; }
  const double s = 1.0 / sqrt(2.0);
  for (int m = -l; m < 0; ++m) { qr[l+m][l-m] = s; qi[l+m][l+m] = -s; }
  qr[l][l] = 1.0;
  for (int m = 1; m <= l; ++m) { double sg = (m & 1) ? -1.0 : 1.0; qr[l+m][l+m] = sg*s; qi[l+m][l-m] = sg*s; }
  if (l == 1) {        // * (-i)
    for (int a = 0; a < 5; ++a) for (int b = 0; b < 5; ++b) { double re = qr[a][b]; qr[a][b] = qi[a][b]; qi[a][b] = -re; }
  } else if (l == 2) { // * (-1)
    for (int a = 0; a < 5; ++a) for (int b = 0; b < 5; ++b) { qr[a][b] = -qr[a][b]; qi[a][b] = -qi[a][b]; }
  }
}
__global__ void w3j_init(float* w3j) {
  const int keys[11][3] = {{0,0,0},{0,1,1},{0,2,2},{1,0,1},{1,1,0},{1,1,2},{1,2,1},{2,0,2},{2,1,1},{2,2,0},{2,2,2}};
  int t = threadIdx.x;
  if (t >= 11) return;
  int l1 = keys[t][0], l2 = keys[t][1], l3 = keys[t][2];
  int d1 = 2*l1+1, d2 = 2*l2+1, d3 = 2*l3+1;
  double q1r[5][5], q1i[5][5], q2r[5][5], q2i[5][5], q3r[5][5], q3i[5][5];
  qmat(l1,q1r,q1i); qmat(l2,q2r,q2i); qmat(l3,q3r,q3i);
  double Tre[125], Tim[125]; double nr = 0, ni = 0;
  for (int a = 0; a < d1; ++a) for (int b = 0; b < d2; ++b) for (int c = 0; c < d3; ++c) {
    double sre = 0, sim = 0;
    for (int i = 0; i < d1; ++i) for (int j = 0; j < d2; ++j) {
      double pr = q1r[a][i]*q2r[b][j] - q1i[a][i]*q2i[b][j];
      double pi = q1r[a][i]*q2i[b][j] + q1i[a][i]*q2r[b][j];
      if (pr == 0 && pi == 0) continue;
      for (int k = 0; k < d3; ++k) {
        double cg = su2cg(l1, i-l1, l2, j-l2, l3, k-l3);
        if (cg == 0) continue;
        double cr = q3r[c][k], ci = -q3i[c][k];
        sre += (pr*cr - pi*ci) * cg;
        sim += (pr*ci + pi*cr) * cg;
      }
    }
    Tre[(a*d2+b)*d3+c] = sre; Tim[(a*d2+b)*d3+c] = sim;
    nr += sre*sre; ni += sim*sim;
  }
  bool useRe = nr >= ni;
  double nrm = sqrt(useRe ? nr : ni);
  float* dst = w3j + t * 128;
  int tot = d1*d2*d3;
  for (int x = 0; x < tot; ++x) dst[x] = (float)((useRe ? Tre[x] : Tim[x]) / nrm);
  for (int x = tot; x < 128; ++x) dst[x] = 0.f;
}

// ---------------------------------------------------------------------------
// Edge geometry: spherical harmonics (l<=2) + soft-one-hot radial basis.
// ---------------------------------------------------------------------------
__global__ void edge_geom(const float* pos, const int* row, const int* col,
                          float* sh, float* emb) {
  int e = blockIdx.x * blockDim.x + threadIdx.x;
  if (e >= E_N) return;
  int r = row[e], c = col[e];
  float ex = pos[r*3+0]-pos[c*3+0], ey = pos[r*3+1]-pos[c*3+1], ez = pos[r*3+2]-pos[c*3+2];
  float d  = sqrtf(ex*ex + ey*ey + ez*ez);
  float dn = fmaxf(d, 1e-9f);
  float x = ex/dn, y = ey/dn, z = ez/dn;
  const float s3 = 1.7320508075688772f, s5 = 2.2360679774997896f, s15 = 3.872983346207417f;
  float* S = sh + (size_t)e * 9;
  S[0] = 1.f; S[1] = s3*x; S[2] = s3*y; S[3] = s3*z;
  S[4] = s15*x*y; S[5] = s15*y*z; S[6] = 0.5f*s5*(3.f*z*z-1.f);
  S[7] = s15*x*z; S[8] = 0.5f*s15*(x*x-y*y);
  float* Eb = emb + (size_t)e * 16;
  const float step = 5.0f / 17.0f;
  for (int i = 0; i < 16; ++i) {
    float v = step * (i + 1);
    float diff = (d - v) / step;
    float val = 0.f;
    if (fabsf(diff) < 1.f) val = 1.14136f * expf(2.f - 2.f / (1.f - diff*diff));
    Eb[i] = val * 4.f;   // * sqrt(NUM_BASIS)
  }
}

// ---------------------------------------------------------------------------
// Equivariant tensor product + scatter-sum (one thread per edge).
// ---------------------------------------------------------------------------
__device__ inline int w3j_id(int a, int b, int c) {
  if (a == 0) { if (b == 0) return 0; if (b == 1) return 1; return 2; }
  if (a == 1) { if (b == 0) return 3; if (b == 1) return (c == 0) ? 4 : 5; return 6; }
  if (b == 0) return 7; if (b == 1) return 8; return (c == 0) ? 9 : 10;
}
__global__ __launch_bounds__(256) void tp_scatter(const float* src, const int* row,
    const int* col, const float* sh, const float* fcw, const float* w3j, float* out) {
  __shared__ float W[11 * 128];
  for (int i = threadIdx.x; i < 11 * 128; i += blockDim.x) W[i] = w3j[i];
  __syncthreads();
  long long e = (long long)blockIdx.x * blockDim.x + threadIdx.x;
  if (e >= E_N) return;
  const float* fr = src + (size_t)row[e] * 162;
  const float* S  = sh  + (size_t)e * 9;
  const float* wv = fcw + (size_t)e * 140;
  float sm[108];
  for (int i = 0; i < 108; ++i) sm[i] = 0.f;
  const int g_lin[3] = {0,1,2}, g_mul[3] = {1,3,5}, g_i0[3] = {0,1,4};
  const int g_wOff[3] = {0,12,60}, g_np[3] = {3,4,4};
  const int g_pl2[3][4] = {{0,1,2,0},{0,1,1,2},{0,1,2,2}};
  const int g_po [3][4] = {{0,1,2,0},{1,0,2,1},{2,1,0,2}};
  const float g_fan[3][3] = {{1,1,1},{3,6,3},{5,5,10}};
  const int oOff[3] = {0,4,16};
  float xm[5][5], t1[5][5], t2[5][5];
  for (int gg = 0; gg < 3; ++gg) {
    int l1 = g_lin[gg], mul = g_mul[gg], i0 = g_i0[gg], d1 = 2*l1+1;
    for (int u = 0; u < mul; ++u)
      for (int i = 0; i < d1; ++i)
        xm[u][i] = fr[(i0+u)*18 + (i0+i)*2 + 0] + fr[(i0+u)*18 + (i0+i)*2 + 1];
    int pOff = g_wOff[gg];
    for (int p = 0; p < g_np[gg]; ++p) {
      int l2 = g_pl2[gg][p], o = g_po[gg][p], l3 = o;  // OUT_LS = (0,1,2)
      int d2 = 2*l2+1, d3 = 2*l3+1;
      const float* C = W + w3j_id(l1, l2, l3) * 128;
      for (int i = 0; i < d1; ++i) for (int k = 0; k < d3; ++k) {
        float s = 0.f;
        for (int j = 0; j < d2; ++j) s += S[l2*l2 + j] * C[(i*d2 + j)*d3 + k];
        t1[i][k] = s;
      }
      for (int u = 0; u < mul; ++u) for (int k = 0; k < d3; ++k) {
        float s = 0.f;
        for (int i = 0; i < d1; ++i) s += xm[u][i] * t1[i][k];
        t2[u][k] = s;
      }
      float pw = sqrtf((float)d3 / g_fan[gg][o]);
      const float* wp = wv + pOff;
      float* op = sm + gg*36 + oOff[o];
      for (int w = 0; w < 4; ++w) for (int k = 0; k < d3; ++k) {
        float s = 0.f;
        for (int u = 0; u < mul; ++u) s += wp[u*4 + w] * t2[u][k];
        op[w*d3 + k] += pw * s;
      }
      pOff += mul * 4;
    }
  }
  float* dst = out + (size_t)col[e] * 108;
  for (int i = 0; i < 108; ++i) atomicAdd(&dst[i], sm[i]);
}

// ---------------------------------------------------------------------------
// Small utility kernels.
// ---------------------------------------------------------------------------
__global__ void seg_countf(const int* idx, long long n, float* cnt) {
  long long i = (long long)blockIdx.x * blockDim.x + threadIdx.x;
  if (i < n) atomicAdd(&cnt[idx[i]], 1.f);
}
__global__ void compose_idx(const int* a, const int* b, int* out, long long n) {
  long long i = (long long)blockIdx.x * blockDim.x + threadIdx.x;
  if (i < n) out[i] = a[b[i]];
}
__global__ void seg_accf(const float* v, const int* idx, long long rows, int F, float* out) {
  long long i = (long long)blockIdx.x * blockDim.x + threadIdx.x;
  if (i >= rows * F) return;
  long long r = i / F; int f = (int)(i % F);
  atomicAdd(&out[(long long)idx[r] * F + f], v[r * F + f]);
}
__global__ void div_rows(float* o, const float* cnt, long long rows, int F) {
  long long i = (long long)blockIdx.x * blockDim.x + threadIdx.x;
  if (i >= rows * F) return;
  o[i] /= fmaxf(cnt[i / F], 1.f);
}
__global__ void scale_buf(float* p, long long n, float s) {
  long long i = (long long)blockIdx.x * blockDim.x + threadIdx.x;
  if (i < n) p[i] *= s;
}

// ---------------------------------------------------------------------------
// Host orchestration.
// ---------------------------------------------------------------------------
extern "C" void kernel_launch(void* const* d_in, const int* in_sizes, int n_in,
                              void* d_out, int out_size, void* d_ws, size_t ws_size,
                              hipStream_t stream) {
  (void)in_sizes; (void)n_in; (void)out_size; (void)ws_size;
  // Top-level dict order: x, edge_attr, u, pos, params(pytree, alphabetical),
  // edge_index, batch, num_nodes.  Dict leaves alphabetical: b before w, etc.
  const float* in_x   = (const float*)d_in[0];
  const float* in_ea  = (const float*)d_in[1];
  const float* in_u   = (const float*)d_in[2];
  const float* in_pos = (const float*)d_in[3];
  const float* edge_l1b = (const float*)d_in[4];
  const float* edge_l1w = (const float*)d_in[5];
  const float* edge_l2b = (const float*)d_in[6];
  const float* edge_l2w = (const float*)d_in[7];
  const float* eg_l1b = (const float*)d_in[14];
  const float* eg_l1w = (const float*)d_in[15];
  const float* eg_l2b = (const float*)d_in[16];
  const float* eg_l2w = (const float*)d_in[17];
  const float* gl_l1b = (const float*)d_in[24];
  const float* gl_l1w = (const float*)d_in[25];
  const float* gl_l2b = (const float*)d_in[26];
  const float* gl_l2w = (const float*)d_in[27];
  const float* n1_l1b = (const float*)d_in[28];
  const float* n1_l1w = (const float*)d_in[29];
  const float* n1_l2b = (const float*)d_in[30];
  const float* n1_l2w = (const float*)d_in[31];
  const float* n2_l1b = (const float*)d_in[32];
  const float* n2_l1w = (const float*)d_in[33];
  const float* n2_l2b = (const float*)d_in[34];
  const float* n2_l2w = (const float*)d_in[35];
  // fc nets: [conv 0=node(equi_node), 1=edge(equi_edge)][group 0=s,1=p,2=d]
  const float* fcw1[2][3] = {
    {(const float*)d_in[22], (const float*)d_in[20], (const float*)d_in[18]},
    {(const float*)d_in[12], (const float*)d_in[10], (const float*)d_in[8]}};
  const float* fcw2[2][3] = {
    {(const float*)d_in[23], (const float*)d_in[21], (const float*)d_in[19]},
    {(const float*)d_in[13], (const float*)d_in[11], (const float*)d_in[9]}};
  const int* eidx  = (const int*)d_in[36];
  const int* batch = (const int*)d_in[37];
  const int* row = eidx;
  const int* col = eidx + E_N;

  float* outX  = (float*)d_out;
  float* outEA = outX + (size_t)N_N * DTP;
  float* outU  = outEA + (size_t)E_N * DTP;

  // Workspace layout
  char* base = (char*)d_ws;
  size_t off = 0;
  auto alloc = [&](size_t bytes) -> size_t {
    size_t r = (off + 255) & ~(size_t)255; off = r + bytes; return r;
  };
  size_t o_w3j  = alloc(11 * 128 * 4);
  size_t o_bcnt = alloc(G_N * 4);
  size_t o_ccnt = alloc((size_t)N_N * 4);
  size_t o_bre  = alloc((size_t)E_N * 4);
  size_t o_sh   = alloc((size_t)E_N * 9 * 4);
  size_t o_emb  = alloc((size_t)E_N * 16 * 4);
  size_t o_gx   = alloc(64 * 108 * 4);
  size_t o_hidg = alloc(64 * 128 * 4);
  size_t o_mh   = alloc((size_t)N_N * 256 * 4);
  // packed bf16 weights (contiguous region, memset once); Npad multiple of 64
  size_t o_wt_fc1[2][3], o_wt_fc2[2][3];
  const int numel[3] = {12, 48, 80}, npad2[3] = {64, 64, 128}, wOffG[3] = {0, 12, 60};
  size_t wtStart = (off + 255) & ~(size_t)255;
  for (int cv = 0; cv < 2; ++cv) for (int g = 0; g < 3; ++g) o_wt_fc1[cv][g] = alloc(128 * 32 * 2);
  for (int cv = 0; cv < 2; ++cv) for (int g = 0; g < 3; ++g) o_wt_fc2[cv][g] = alloc((size_t)npad2[g] * 128 * 2);
  size_t o_wt_eg1 = alloc(128 * 160 * 2), o_wt_eg2 = alloc(128 * 128 * 2);
  size_t o_wt_ed1 = alloc(256 * 480 * 2), o_wt_ed2 = alloc(128 * 256 * 2);
  size_t o_wt_n11 = alloc(256 * 224 * 2), o_wt_n12 = alloc(256 * 256 * 2);
  size_t o_wt_n21 = alloc(256 * 512 * 2), o_wt_n22 = alloc(128 * 256 * 2);
  size_t o_wt_gl1 = alloc(128 * 256 * 2), o_wt_gl2 = alloc(128 * 128 * 2);
  size_t wtEnd = off;
  size_t o_bb0 = alloc((size_t)E_N * 256 * 4);   // hidden activations
  size_t o_bb1 = alloc((size_t)E_N * 256 * 4);   // fcw (E x 140) / h (E x 256)

  float* w3j  = (float*)(base + o_w3j);
  float* bcnt = (float*)(base + o_bcnt);
  float* ccnt = (float*)(base + o_ccnt);
  int*   bre  = (int*)(base + o_bre);
  float* shb  = (float*)(base + o_sh);
  float* emb  = (float*)(base + o_emb);
  float* gx   = (float*)(base + o_gx);
  float* hidg = (float*)(base + o_hidg);
  float* mh   = (float*)(base + o_mh);
  float* bb0  = (float*)(base + o_bb0);
  float* bb1  = (float*)(base + o_bb1);

  auto rungemm = [&](GemmP& p) {
    long long T = (p.M / 16) * (long long)(p.Npad / 64);
    int blocks = (int)((T + 7) / 8);
    gemm_bf16_wmma<<<blocks, 256, 0, stream>>>(p);
  };
  auto pack = [&](const float* W, size_t wtOff, int K, int N, int Kpad) {
    int total = N * Kpad;
    pack_wt<<<(total + 255) / 256, 256, 0, stream>>>(
        W, reinterpret_cast<__bf16*>(base + wtOff), K, N, Kpad);
  };

  // ---- init ----
  hipMemsetAsync(base + o_bcnt, 0, G_N * 4, stream);
  hipMemsetAsync(base + o_ccnt, 0, (size_t)N_N * 4, stream);
  hipMemsetAsync(base + wtStart, 0, wtEnd - wtStart, stream);
  w3j_init<<<1, 32, 0, stream>>>(w3j);
  seg_countf<<<N_N / 256, 256, 0, stream>>>(batch, N_N, bcnt);
  seg_countf<<<E_N / 256, 256, 0, stream>>>(col, E_N, ccnt);
  compose_idx<<<E_N / 256, 256, 0, stream>>>(batch, row, bre, E_N);
  edge_geom<<<E_N / 256, 256, 0, stream>>>(in_pos, row, col, shb, emb);

  for (int cv = 0; cv < 2; ++cv) for (int g = 0; g < 3; ++g) {
    pack(fcw1[cv][g], o_wt_fc1[cv][g], 16, 128, 32);
    pack(fcw2[cv][g], o_wt_fc2[cv][g], 128, numel[g], 128);
  }
  pack(eg_l1w, o_wt_eg1, 140, 128, 160);  pack(eg_l2w, o_wt_eg2, 128, 128, 128);
  pack(edge_l1w, o_wt_ed1, 452, 256, 480); pack(edge_l2w, o_wt_ed2, 256, 108, 256);
  pack(n1_l1w, o_wt_n11, 216, 256, 224);  pack(n1_l2w, o_wt_n12, 256, 256, 256);
  pack(n2_l1w, o_wt_n21, 492, 256, 512);  pack(n2_l2w, o_wt_n22, 256, 108, 256);
  pack(gl_l1w, o_wt_gl1, 236, 128, 256);  pack(gl_l2w, o_wt_gl2, 128, 128, 128);

  // ---- equivariant convolutions ----
  for (int cv = 0; cv < 2; ++cv) {
    for (int g = 0; g < 3; ++g) {
      GemmP p1{}; p1.seg[0] = {emb, nullptr, 16}; p1.nseg = 1;
      p1.M = E_N; p1.K = 16; p1.Kpad = 32; p1.N = 128; p1.Npad = 128;
      p1.WT = (const __bf16*)(base + o_wt_fc1[cv][g]); p1.bias = nullptr;
      p1.alpha = 0.25f; p1.postscale = 1.41421356f; p1.relu = 1;
      p1.C = bb0; p1.ldC = 128; p1.colOff = 0;
      rungemm(p1);
      GemmP p2{}; p2.seg[0] = {bb0, nullptr, 128}; p2.nseg = 1;
      p2.M = E_N; p2.K = 128; p2.Kpad = 128; p2.N = numel[g]; p2.Npad = npad2[g];
      p2.WT = (const __bf16*)(base + o_wt_fc2[cv][g]); p2.bias = nullptr;
      p2.alpha = 0.0883883476f; p2.postscale = 1.f; p2.relu = 0;
      p2.C = bb1; p2.ldC = 140; p2.colOff = wOffG[g];
      rungemm(p2);
    }
    const float* src = (cv == 0) ? in_x : in_ea;
    float* dst = (cv == 0) ? outX : outEA;
    long long dlen = ((cv == 0) ? (long long)N_N : (long long)E_N) * DTP;
    hipMemsetAsync(dst, 0, (size_t)dlen * 4, stream);
    tp_scatter<<<E_N / 256, 256, 0, stream>>>(src, row, col, shb, bb1, w3j, dst);
    scale_buf<<<(int)((dlen + 255) / 256), 256, 0, stream>>>(dst, dlen, 0.25f);
  }

  // ---- global MLP ----
  hipMemsetAsync(gx, 0, 64 * 108 * 4, stream);
  seg_accf<<<(N_N * 108 + 255) / 256, 256, 0, stream>>>(outX, batch, N_N, 108, gx);
  div_rows<<<(64 * 108 + 255) / 256, 256, 0, stream>>>(gx, bcnt, 64, 108);
  {
    GemmP p{}; p.seg[0] = {in_u, nullptr, 32}; p.seg[1] = {gx, nullptr, 108}; p.nseg = 2;
    p.M = 64; p.K = 140; p.Kpad = 160; p.N = 128; p.Npad = 128;
    p.WT = (const __bf16*)(base + o_wt_eg1); p.bias = eg_l1b;
    p.alpha = 1.f; p.postscale = 1.f; p.relu = 1; p.C = hidg; p.ldC = 128; p.colOff = 0;
    rungemm(p);
    GemmP q{}; q.seg[0] = {hidg, nullptr, 128}; q.nseg = 1;
    q.M = 64; q.K = 128; q.Kpad = 128; q.N = 128; q.Npad = 128;
    q.WT = (const __bf16*)(base + o_wt_eg2); q.bias = eg_l2b;
    q.alpha = 1.f; q.postscale = 1.f; q.relu = 0; q.C = outU; q.ldC = 128; q.colOff = 0;
    rungemm(q);
  }

  // ---- message-passing updates ----
  for (int it = 0; it < 2; ++it) {
    GemmP e1{};
    e1.seg[0] = {outX, row, 108}; e1.seg[1] = {outX, col, 108};
    e1.seg[2] = {outEA, nullptr, 108}; e1.seg[3] = {outU, bre, 128}; e1.nseg = 4;
    e1.M = E_N; e1.K = 452; e1.Kpad = 480; e1.N = 256; e1.Npad = 256;
    e1.WT = (const __bf16*)(base + o_wt_ed1); e1.bias = edge_l1b;
    e1.alpha = 1.f; e1.postscale = 1.f; e1.relu = 1; e1.C = bb0; e1.ldC = 256; e1.colOff = 0;
    rungemm(e1);
    GemmP e2{}; e2.seg[0] = {bb0, nullptr, 256}; e2.nseg = 1;
    e2.M = E_N; e2.K = 256; e2.Kpad = 256; e2.N = 108; e2.Npad = 128;
    e2.WT = (const __bf16*)(base + o_wt_ed2); e2.bias = edge_l2b;
    e2.alpha = 1.f; e2.postscale = 1.f; e2.relu = 0; e2.C = outEA; e2.ldC = 108; e2.colOff = 0;
    rungemm(e2);

    GemmP a1{}; a1.seg[0] = {outX, row, 108}; a1.seg[1] = {outEA, nullptr, 108}; a1.nseg = 2;
    a1.M = E_N; a1.K = 216; a1.Kpad = 224; a1.N = 256; a1.Npad = 256;
    a1.WT = (const __bf16*)(base + o_wt_n11); a1.bias = n1_l1b;
    a1.alpha = 1.f; a1.postscale = 1.f; a1.relu = 1; a1.C = bb0; a1.ldC = 256; a1.colOff = 0;
    rungemm(a1);
    GemmP a2{}; a2.seg[0] = {bb0, nullptr, 256}; a2.nseg = 1;
    a2.M = E_N; a2.K = 256; a2.Kpad = 256; a2.N = 256; a2.Npad = 256;
    a2.WT = (const __bf16*)(base + o_wt_n12); a2.bias = n1_l2b;
    a2.alpha = 1.f; a2.postscale = 1.f; a2.relu = 0; a2.C = bb1; a2.ldC = 256; a2.colOff = 0;
    rungemm(a2);

    hipMemsetAsync(mh, 0, (size_t)N_N * 256 * 4, stream);
    seg_accf<<<(int)(((long long)E_N * 256 + 255) / 256), 256, 0, stream>>>(bb1, col, E_N, 256, mh);
    div_rows<<<(int)(((long long)N_N * 256 + 255) / 256), 256, 0, stream>>>(mh, ccnt, N_N, 256);

    GemmP b1{};
    b1.seg[0] = {outX, nullptr, 108}; b1.seg[1] = {mh, nullptr, 256};
    b1.seg[2] = {outU, batch, 128}; b1.nseg = 3;
    b1.M = N_N; b1.K = 492; b1.Kpad = 512; b1.N = 256; b1.Npad = 256;
    b1.WT = (const __bf16*)(base + o_wt_n21); b1.bias = n2_l1b;
    b1.alpha = 1.f; b1.postscale = 1.f; b1.relu = 1; b1.C = bb0; b1.ldC = 256; b1.colOff = 0;
    rungemm(b1);
    GemmP b2{}; b2.seg[0] = {bb0, nullptr, 256}; b2.nseg = 1;
    b2.M = N_N; b2.K = 256; b2.Kpad = 256; b2.N = 108; b2.Npad = 128;
    b2.WT = (const __bf16*)(base + o_wt_n22); b2.bias = n2_l2b;
    b2.alpha = 1.f; b2.postscale = 1.f; b2.relu = 0; b2.C = outX; b2.ldC = 108; b2.colOff = 0;
    rungemm(b2);

    hipMemsetAsync(gx, 0, 64 * 108 * 4, stream);
    seg_accf<<<(N_N * 108 + 255) / 256, 256, 0, stream>>>(outX, batch, N_N, 108, gx);
    div_rows<<<(64 * 108 + 255) / 256, 256, 0, stream>>>(gx, bcnt, 64, 108);
    GemmP g1{}; g1.seg[0] = {outU, nullptr, 128}; g1.seg[1] = {gx, nullptr, 108}; g1.nseg = 2;
    g1.M = 64; g1.K = 236; g1.Kpad = 256; g1.N = 128; g1.Npad = 128;
    g1.WT = (const __bf16*)(base + o_wt_gl1); g1.bias = gl_l1b;
    g1.alpha = 1.f; g1.postscale = 1.f; g1.relu = 1; g1.C = hidg; g1.ldC = 128; g1.colOff = 0;
    rungemm(g1);
    GemmP g2{}; g2.seg[0] = {hidg, nullptr, 128}; g2.nseg = 1;
    g2.M = 64; g2.K = 128; g2.Kpad = 128; g2.N = 128; g2.Npad = 128;
    g2.WT = (const __bf16*)(base + o_wt_gl2); g2.bias = gl_l2b;
    g2.alpha = 1.f; g2.postscale = 1.f; g2.relu = 0; g2.C = outU; g2.ldC = 128; g2.colOff = 0;
    rungemm(g2);
  }
}
